// VCTree_GCL_88871463289481
// MI455X (gfx1250) — matrix operand order repaired and checked by reading, not compiled, not run on hardware
//
#include <hip/hip_runtime.h>
#include <hip/hip_bf16.h>

// ---------------------------------------------------------------------------
// Problem constants (from reference)
// ---------------------------------------------------------------------------
#define N_OBJ  1280
#define HID    512
#define N_REL  65536
#define POOL   4096
#define MLPD   2048
#define EMB2   (2 * HID)      // 1024

typedef unsigned short ushort_t;
typedef unsigned int   uint_t;

typedef __attribute__((ext_vector_type(16))) __bf16 v16bf;
typedef __attribute__((ext_vector_type(2)))  __bf16 v2bf;
typedef __attribute__((ext_vector_type(8)))  float  v8f;
typedef __attribute__((ext_vector_type(4)))  unsigned int u32x4;

union FragU {
    v16bf v;
    u32x4 q[2];
};

// f32 -> bf16 (round half up), scalar
__device__ __forceinline__ uint_t f2bf_u(float f) {
    return (__float_as_uint(f) + 0x8000u) >> 16;
}

// pack two f32 into one u32 holding two bf16 (a = low half, b = high half)
#if __has_builtin(__builtin_amdgcn_cvt_pk_bf16_f32)
__device__ __forceinline__ uint_t pk2bf(float a, float b) {
    union { v2bf v; uint_t u; } c;
    c.v = __builtin_amdgcn_cvt_pk_bf16_f32(a, b);
    return c.u;
}
#else
// round-half-up then one v_perm_b32 picks top bytes of both: 3 VALU per pair
__device__ __forceinline__ uint_t pk2bf(float a, float b) {
    uint_t ua = __float_as_uint(a) + 0x8000u;
    uint_t ub = __float_as_uint(b) + 0x8000u;
    // dst = { ub[31:24], ub[23:16], ua[31:24], ua[23:16] }
    return __builtin_amdgcn_perm(ub, ua, 0x07060302u);
}
#endif

__device__ __forceinline__ v8f wmma_bf16(v16bf a, v16bf b, v8f c) {
    return __builtin_amdgcn_wmma_f32_16x16x32_bf16(
        false, a, false, b, (short)0, c, false, false);
}

// A fragment: 16x32 bf16 from LDS row-major [128][32]  (contiguous-k order, matched by B)
// lane L: row M = rowbase + (L&15); kb = (L>>4)*16
// halves 0..15 = K kb..kb+15 (contiguous)
__device__ __forceinline__ v16bf load_fragA(const ushort_t* ldsA, int rowbase, int lane) {
    int r  = rowbase + (lane & 15);
    int kb = (lane >> 4) * 16;
    FragU f;
    f.q[0] = *(const u32x4*)(ldsA + r * 32 + kb);
    f.q[1] = *(const u32x4*)(ldsA + r * 32 + kb + 8);
    return f.v;
}

// B fragment: 32x16 bf16 loaded straight from global pre-transposed Wt[n][k]
// lane L: col N = n (absolute); halves 0..15 = K kb..kb+15 (contiguous, 32B)
__device__ __forceinline__ v16bf load_fragB_g(const ushort_t* __restrict__ Wt,
                                              int Kdim, int n, int kb) {
    const ushort_t* p = Wt + (size_t)n * Kdim + kb;
    FragU f;
    f.q[0] = *(const u32x4*)(p);
    f.q[1] = *(const u32x4*)(p + 8);
    return f.v;
}

// stage 16 consecutive f32 -> 16 bf16 (packed) -> 2x ds_store_b128
__device__ __forceinline__ void stage16cvt(const float* __restrict__ p, ushort_t* dst) {
    float4 f0 = ((const float4*)p)[0];
    float4 f1 = ((const float4*)p)[1];
    float4 f2 = ((const float4*)p)[2];
    float4 f3 = ((const float4*)p)[3];
    u32x4 q0 = { pk2bf(f0.x, f0.y), pk2bf(f0.z, f0.w),
                 pk2bf(f1.x, f1.y), pk2bf(f1.z, f1.w) };
    u32x4 q1 = { pk2bf(f2.x, f2.y), pk2bf(f2.z, f2.w),
                 pk2bf(f3.x, f3.y), pk2bf(f3.z, f3.w) };
    *(u32x4*)(dst)     = q0;
    *(u32x4*)(dst + 8) = q1;
}

// pure bf16 copy of 16 elements (global -> LDS), 2x b128 each way
__device__ __forceinline__ void copy16bf(const ushort_t* __restrict__ src, ushort_t* dst) {
    u32x4 q0 = ((const u32x4*)src)[0];
    u32x4 q1 = ((const u32x4*)src)[1];
    *(u32x4*)(dst)     = q0;
    *(u32x4*)(dst + 8) = q1;
}

// one K-step of the 128x128 tile GEMM for one wave (2 row frags x 4 col frags)
__device__ __forceinline__ void gemm_step(
    const ushort_t* __restrict__ cur,
    const ushort_t* __restrict__ Wt, int Kdim,
    int k0, int nbase, int rowb, int lane, v8f (&acc)[2][4])
{
    v16bf aF0 = load_fragA(cur, rowb, lane);
    v16bf aF1 = load_fragA(cur, rowb + 16, lane);
    const int kb = k0 + (lane >> 4) * 16;
#pragma unroll
    for (int fn = 0; fn < 4; ++fn) {
        v16bf bF = load_fragB_g(Wt, Kdim, nbase + fn * 16, kb);
        acc[0][fn] = wmma_bf16(aF0, bF, acc[0][fn]);
        acc[1][fn] = wmma_bf16(aF1, bF, acc[1][fn]);
    }
}

// ---------------------------------------------------------------------------
// One-time: W[k][n] f32  ->  Wt[n][k] bf16 (fragment-friendly layout)
// grid (K/64, N/64), block 256, LDS 64x64 f32 tile (padded stride 65)
// ---------------------------------------------------------------------------
__global__ __launch_bounds__(256) void transpose_bf16_kernel(
    const float* __restrict__ W, ushort_t* __restrict__ Wt, int K, int N)
{
    __shared__ float tile[64 * 65];
    const int k0 = blockIdx.x * 64;
    const int n0 = blockIdx.y * 64;
    const int t  = threadIdx.x;
    {
        int kl = t >> 2, c = (t & 3) * 16;
        const float* p = W + (size_t)(k0 + kl) * N + n0 + c;
#pragma unroll
        for (int j = 0; j < 16; ++j) tile[kl * 65 + c + j] = p[j];
    }
    __syncthreads();
    {
        int nl = t >> 2, c = (t & 3) * 16;   // c = local-k segment
        uint_t w[8];
#pragma unroll
        for (int j = 0; j < 8; ++j)
            w[j] = pk2bf(tile[(c + 2 * j) * 65 + nl],
                         tile[(c + 2 * j + 1) * 65 + nl]);
        ushort_t* dst = Wt + (size_t)(n0 + nl) * K + k0 + c;
        *(u32x4*)(dst)     = (u32x4){ w[0], w[1], w[2], w[3] };
        *(u32x4*)(dst + 8) = (u32x4){ w[4], w[5], w[6], w[7] };
    }
}

// ---------------------------------------------------------------------------
// edge_rep = relu(edge_ctx @ W_post_emb + b)  -> bf16 [1280][1024]
// grid (8, 10): x = col tile, y = row tile; 128x128 tiles; K = 512
// ---------------------------------------------------------------------------
__global__ __launch_bounds__(256) void edge_rep_kernel(
    const float* __restrict__ edge_ctx,   // [1280][512] f32
    const ushort_t* __restrict__ Wt_emb,  // [1024][512] bf16
    const float* __restrict__ b_post_emb, // [1024]
    ushort_t* __restrict__ edge_bf)       // [1280][1024] bf16 out
{
    __shared__ ushort_t ldsA[2][128 * 32];
    const int tid   = threadIdx.x;
    const int lane  = tid & 31;
    const int wid   = tid >> 5;
    const int n0    = blockIdx.x * 128;
    const int row0  = blockIdx.y * 128;
    const int rowb  = (wid & 3) * 32;
    const int colb  = (wid >> 2) * 64;
    const int so    = (tid >> 1) * 32 + (tid & 1) * 16;   // staging LDS offset
    const int nbase = n0 + colb + (lane & 15);

    v8f acc[2][4];
#pragma unroll
    for (int i = 0; i < 2; ++i)
#pragma unroll
        for (int j = 0; j < 4; ++j) acc[i][j] = (v8f)0.0f;

    const float* aSrc = edge_ctx + (size_t)(row0 + (tid >> 1)) * HID + (tid & 1) * 16;
    stage16cvt(aSrc, &ldsA[0][so]);
    __syncthreads();
    int pb = 0;
    for (int k0 = 0; k0 < HID - 32; k0 += 32) {
        stage16cvt(aSrc + k0 + 32, &ldsA[pb ^ 1][so]);
        gemm_step(ldsA[pb], Wt_emb, HID, k0, nbase, rowb, lane, acc);
        __syncthreads();
        pb ^= 1;
    }
    gemm_step(ldsA[pb], Wt_emb, HID, HID - 32, nbase, rowb, lane, acc);

    const int mofs = (lane >> 4) * 8;
#pragma unroll
    for (int fm = 0; fm < 2; ++fm) {
#pragma unroll
        for (int fn = 0; fn < 4; ++fn) {
            int gn = nbase + fn * 16;
            float bias = b_post_emb[gn];
#pragma unroll
            for (int r = 0; r < 8; ++r) {
                int gm = row0 + rowb + fm * 16 + mofs + r;
                float v = acc[fm][fn][r] + bias;
                v = v > 0.0f ? v : 0.0f;
                edge_bf[(size_t)gm * EMB2 + gn] = (ushort_t)f2bf_u(v);
            }
        }
    }
}

// ---------------------------------------------------------------------------
// Fused main: per 128x128 output tile
//   U = union_features @ W_up            (K=2048, A f32->bf16 staged to LDS)
//   P = gather(edge_bf) @ W_post_cat     (K=1024, A pure bf16 gather-copy;
//       row selector flips from pair_idx[:,0] to [:,1] exactly at k=512,
//       so the loop is split there -> straight-line bodies, no branches)
//   out = (P + b_post_cat) * (U + b_up)   [non-temporal stores]
// grid (32, 512): x = col tile (fast) so the 32 concurrent blocks share each
// union A-tile in L2 -> union_features pulled from HBM ~once (512 MB).
// ---------------------------------------------------------------------------
__global__ __launch_bounds__(256) void fused_main_kernel(
    const ushort_t* __restrict__ edge_bf,     // [1280][1024] bf16 (ws)
    const long long* __restrict__ pair_idx,   // [65536][2] int64
    const float* __restrict__ union_features, // [65536][2048] f32
    const ushort_t* __restrict__ Wt_cat,      // [4096][1024] bf16 (ws)
    const float* __restrict__ b_post_cat,     // [4096]
    const ushort_t* __restrict__ Wt_up,       // [4096][2048] bf16 (ws)
    const float* __restrict__ b_up,           // [4096]
    float* __restrict__ out)                  // [65536][4096]
{
    __shared__ ushort_t ldsA[2][128 * 32];
    const int tid   = threadIdx.x;
    const int lane  = tid & 31;
    const int wid   = tid >> 5;
    const int n0    = blockIdx.x * 128;
    const int row0  = blockIdx.y * 128;
    const int rowb  = (wid & 3) * 32;
    const int colb  = (wid >> 2) * 64;
    const int sr    = tid >> 1;
    const int sc    = (tid & 1) * 16;
    const int so    = sr * 32 + sc;
    const int nbase = n0 + colb + (lane & 15);

    v8f accU[2][4];
    v8f accP[2][4];
#pragma unroll
    for (int i = 0; i < 2; ++i)
#pragma unroll
        for (int j = 0; j < 4; ++j) { accU[i][j] = (v8f)0.0f; accP[i][j] = (v8f)0.0f; }

    int pb = 0;

    // -------- Pass U: union_features @ W_up, K = 2048 --------
    const float* aU = union_features + (size_t)(row0 + sr) * MLPD + sc;
    stage16cvt(aU, &ldsA[0][so]);
    __syncthreads();
    for (int k0 = 0; k0 < MLPD - 32; k0 += 32) {
        stage16cvt(aU + k0 + 32, &ldsA[pb ^ 1][so]);
        gemm_step(ldsA[pb], Wt_up, MLPD, k0, nbase, rowb, lane, accU);
        __syncthreads();
        pb ^= 1;
    }
    gemm_step(ldsA[pb], Wt_up, MLPD, MLPD - 32, nbase, rowb, lane, accU);
    __syncthreads();
    pb ^= 1;

    // -------- Pass P: gathered prod_rep @ W_post_cat, K = 1024 --------
    // prod_rep[r][k] = edge_rep[pair_idx[r][ k<512 ? 0 : 1 ]][k] (bf16 copy)
    {
        const long long i0 = pair_idx[(size_t)(row0 + sr) * 2 + 0];
        const long long i1 = pair_idx[(size_t)(row0 + sr) * 2 + 1];
        const ushort_t* aP0 = edge_bf + (size_t)i0 * EMB2 + sc;
        const ushort_t* aP1 = edge_bf + (size_t)i1 * EMB2 + sc;

        copy16bf(aP0, &ldsA[pb][so]);
        __syncthreads();
        // head half: K = 0..480, next tile still from aP0
        for (int k0 = 0; k0 < HID - 32; k0 += 32) {
            copy16bf(aP0 + k0 + 32, &ldsA[pb ^ 1][so]);
            gemm_step(ldsA[pb], Wt_cat, EMB2, k0, nbase, rowb, lane, accP);
            __syncthreads();
            pb ^= 1;
        }
        // bridge step: compute K=480 tile, stage first aP1 tile (k=512)
        copy16bf(aP1 + HID, &ldsA[pb ^ 1][so]);
        gemm_step(ldsA[pb], Wt_cat, EMB2, HID - 32, nbase, rowb, lane, accP);
        __syncthreads();
        pb ^= 1;
        // tail half: K = 512..992, from aP1
        for (int k0 = HID; k0 < EMB2 - 32; k0 += 32) {
            copy16bf(aP1 + k0 + 32, &ldsA[pb ^ 1][so]);
            gemm_step(ldsA[pb], Wt_cat, EMB2, k0, nbase, rowb, lane, accP);
            __syncthreads();
            pb ^= 1;
        }
        gemm_step(ldsA[pb], Wt_cat, EMB2, EMB2 - 32, nbase, rowb, lane, accP);
    }

    // -------- Epilogue: out = (P + b_post_cat) * (U + b_up) --------
    // Non-temporal: output is write-once, keep L2 for the reused weights /
    // union tiles instead of 1 GB of streaming writes.
    const int mofs = (lane >> 4) * 8;
#pragma unroll
    for (int fm = 0; fm < 2; ++fm) {
#pragma unroll
        for (int fn = 0; fn < 4; ++fn) {
            int gn = nbase + fn * 16;
            float bc = b_post_cat[gn];
            float bu = b_up[gn];
#pragma unroll
            for (int r = 0; r < 8; ++r) {
                int gm = row0 + rowb + fm * 16 + mofs + r;
                float v = (accP[fm][fn][r] + bc) * (accU[fm][fn][r] + bu);
                __builtin_nontemporal_store(v, &out[(size_t)gm * POOL + gn]);
            }
        }
    }
}

// ---------------------------------------------------------------------------
// Launch
// ---------------------------------------------------------------------------
extern "C" void kernel_launch(void* const* d_in, const int* in_sizes, int n_in,
                              void* d_out, int out_size, void* d_ws, size_t ws_size,
                              hipStream_t stream) {
    const float*     edge_ctx       = (const float*)d_in[0];
    const long long* pair_idx       = (const long long*)d_in[1];
    const float*     union_features = (const float*)d_in[2];
    const float*     W_post_emb     = (const float*)d_in[3];
    const float*     b_post_emb     = (const float*)d_in[4];
    const float*     W_post_cat     = (const float*)d_in[5];
    const float*     b_post_cat     = (const float*)d_in[6];
    const float*     W_up           = (const float*)d_in[7];
    const float*     b_up           = (const float*)d_in[8];

    // workspace layout (bf16), ~28.8 MB total
    ushort_t* edge_bf = (ushort_t*)d_ws;                // 1280*1024
    ushort_t* Wt_emb  = edge_bf + (size_t)N_OBJ * EMB2; // [1024][512]
    ushort_t* Wt_up   = Wt_emb + (size_t)EMB2 * HID;    // [4096][2048]
    ushort_t* Wt_cat  = Wt_up + (size_t)POOL * MLPD;    // [4096][1024]

    // one-time weight convert+transpose (all L2-resident afterwards)
    transpose_bf16_kernel<<<dim3(HID / 64, EMB2 / 64), 256, 0, stream>>>(
        W_post_emb, Wt_emb, HID, EMB2);
    transpose_bf16_kernel<<<dim3(MLPD / 64, POOL / 64), 256, 0, stream>>>(
        W_up, Wt_up, MLPD, POOL);
    transpose_bf16_kernel<<<dim3(EMB2 / 64, POOL / 64), 256, 0, stream>>>(
        W_post_cat, Wt_cat, EMB2, POOL);

    // tiny GEMM: edge_rep (bf16)
    edge_rep_kernel<<<dim3(EMB2 / 128, N_OBJ / 128), 256, 0, stream>>>(
        edge_ctx, Wt_emb, b_post_emb, edge_bf);

    // fused main GEMMs + elementwise product
    fused_main_kernel<<<dim3(POOL / 128, N_REL / 128), 256, 0, stream>>>(
        edge_bf, pair_idx, union_features, Wt_cat, b_post_cat, Wt_up, b_up,
        (float*)d_out);
}
